// VectorQuantizer_14817637171666
// MI455X (gfx1250) — matrix operand order repaired and checked by reading, not compiled, not run on hardware
//
#include <hip/hip_runtime.h>
#include <math.h>

typedef __attribute__((ext_vector_type(16))) _Float16 v16h;
typedef __attribute__((ext_vector_type(8)))  _Float16 v8h;
typedef __attribute__((ext_vector_type(8)))  float    v8f;

#define KCB   2048           // codebook size
#define DIM   64             // codebook dim
#define BB    32
#define TT    4096
#define NROW  (BB*TT)        // 131072 rows
#define TM    128            // rows per block (one b per block: 128 | 4096)
#define CT    64             // codebook cols staged per LDS iteration
#define RS    72             // LDS row stride in halves (144B: 16B aligned, conflict-spread)
#define NITER (KCB/CT)       // 32 K-chunks
#define WDEC  0.99f

// ---- shared memory byte layout inside vq_argmin ----
//  [0,18432)      xs_hi  (TM*RS halves)  -> reused as cs buf1 (hi 0..9216, lo 9216..18432)
//  [18432,36864)  xs_lo  (TM*RS halves)
//  [36864,46080)  cs buf0 hi  (CT*RS halves)
//  [46080,55296)  cs buf0 lo
#define SMEM_BYTES 55296
#define XSLO_OFF   18432

// ---- workspace byte offsets (all 16B aligned) ----
#define WS_CHI    0                        // K*D halves
#define WS_CLO    (WS_CHI + KCB*DIM*2)     // K*D halves
#define WS_CNORM  (WS_CLO + KCB*DIM*2)     // K floats
#define WS_IDX    (WS_CNORM + KCB*4)       // N ints
#define WS_CBSUM  (WS_IDX + NROW*4)        // K*D floats  (zeroed)
#define WS_COUNTS (WS_CBSUM + KCB*DIM*4)   // K floats    (zeroed)
#define WS_ACCUM  (WS_COUNTS + KCB*4)      // 8 floats    (zeroed)
#define ZERO_FLOATS (KCB*DIM + KCB + 8)

// ---- output float offsets (tuple order) ----
#define OUT_COMMIT (BB*DIM*TT)             // 8388608
#define OUT_FIT    (OUT_COMMIT + 1)
#define OUT_XNORM  (OUT_COMMIT + 2)
#define OUT_NEWCB  (OUT_COMMIT + 3)
#define OUT_NEWAVG (OUT_NEWCB + KCB*DIM)
#define OUT_NEWUSE (OUT_NEWAVG + KCB*DIM)
#define OUT_ENT    (OUT_NEWUSE + KCB)
#define OUT_CBDIFF (OUT_ENT + 1)

// ------------------------------------------------------------------
__global__ void vq_zero(float* __restrict__ p, int n) {
  int i = blockIdx.x * 256 + threadIdx.x;
  if (i < n) p[i] = 0.0f;
}

// ------------------------------------------------------------------
// Split codebook into f16 hi/lo and compute per-codeword squared norms.
__global__ void __launch_bounds__(256) vq_prep_codebook(
    const float* __restrict__ cb, _Float16* __restrict__ chi,
    _Float16* __restrict__ clo, float* __restrict__ cnorm) {
  int wv = threadIdx.x >> 5, lane = threadIdx.x & 31;
  int k = blockIdx.x * 8 + wv;
  float s = 0.0f;
#pragma unroll
  for (int j = 0; j < 2; ++j) {
    int d = lane + j * 32;
    float v = cb[k * DIM + d];
    _Float16 h = (_Float16)v;
    chi[k * DIM + d] = h;
    clo[k * DIM + d] = (_Float16)(v - (float)h);
    s += v * v;
  }
#pragma unroll
  for (int off = 16; off >= 1; off >>= 1) s += __shfl_xor(s, off, 32);
  if (lane == 0) cnorm[k] = s;
}

// ------------------------------------------------------------------
// Main fused distance-GEMM + running argmin kernel (split-f16 WMMA),
// with async-to-LDS double-buffered codebook staging (ASYNCcnt).
__global__ void __launch_bounds__(256) vq_argmin(
    const float* __restrict__ x, const _Float16* __restrict__ chi,
    const _Float16* __restrict__ clo, const float* __restrict__ cnorm,
    int* __restrict__ idx) {
  __shared__ __align__(16) char smem[SMEM_BYTES];
  __shared__ float cns[2][CT];

  _Float16* xs_hi = (_Float16*)smem;
  _Float16* xs_lo = (_Float16*)(smem + XSLO_OFF);
  const unsigned csoff_hi[2] = {36864u, 0u};
  const unsigned csoff_lo[2] = {46080u, 9216u};

  const int tid = threadIdx.x;
  const int lane = tid & 31;
  const int wv = tid >> 5;
  const int n0 = blockIdx.x * TM;     // global row base; whole block in one b
  const int b = n0 / TT;
  const int t0 = n0 % TT;
  const unsigned lds_base = (unsigned)(size_t)(void*)smem;

  // Stage x tile [TM rows][DIM dims] as hi/lo f16 (coalesced over t).
  for (int i = tid; i < TM * DIM; i += 256) {
    int e = i / TM;
    int m = i % TM;
    float vx = x[((size_t)b * DIM + e) * TT + t0 + m];
    _Float16 h = (_Float16)vx;
    xs_hi[m * RS + e] = h;
    xs_lo[m * RS + e] = (_Float16)(vx - (float)h);
  }
  __syncthreads();

  // Cache A fragments in registers for the whole K loop.
  // 16-bit A 16x32 layout: lanes 0-15 -> K {0..7,16..23}; lanes 16-31 -> +8.
  const int mrow = wv * 16 + (lane & 15);
  const int kba = (lane & 16) ? 8 : 0;
  v16h axh[2], axl[2];
#pragma unroll
  for (int kkI = 0; kkI < 2; ++kkI) {
    int kb = kkI * 32 + kba;
    v8h h0 = *(const v8h*)&xs_hi[mrow * RS + kb];
    v8h h1 = *(const v8h*)&xs_hi[mrow * RS + kb + 16];
    v8h l0 = *(const v8h*)&xs_lo[mrow * RS + kb];
    v8h l1 = *(const v8h*)&xs_lo[mrow * RS + kb + 16];
#pragma unroll
    for (int i = 0; i < 8; ++i) {
      axh[kkI][i] = h0[i]; axh[kkI][8 + i] = h1[i];
      axl[kkI][i] = l0[i]; axl[kkI][8 + i] = l1[i];
    }
  }
  // All waves must be done reading xs before buf1 (overlaying xs_hi) is written.
  __syncthreads();

  // Async prefetch of one codebook tile (hi+lo) into LDS buffer `buf`:
  // 4 async b128 instructions per thread => wave ASYNCcnt += 4 per tile.
  auto prefetch = [&](int ctn, int buf) {
#pragma unroll
    for (int i = 0; i < 2; ++i) {
      int c2 = i * 256 + tid;              // chunk id in [0,512)
      int col = c2 >> 3;
      int part = (c2 & 7) * 16;            // byte offset within a 128B row
      unsigned gof = (unsigned)((ctn * CT + col) * (DIM * 2) + part);
      unsigned lofh = lds_base + csoff_hi[buf] + (unsigned)(col * (RS * 2) + part);
      asm volatile("global_load_async_to_lds_b128 %0, %1, %2 offset:0"
                   :: "v"(lofh), "v"(gof), "s"(chi) : "memory");
      unsigned lofl = lds_base + csoff_lo[buf] + (unsigned)(col * (RS * 2) + part);
      asm volatile("global_load_async_to_lds_b128 %0, %1, %2 offset:0"
                   :: "v"(lofl), "v"(gof), "s"(clo) : "memory");
    }
    if (tid < CT) cns[buf][tid] = cnorm[ctn * CT + tid];
  };

  float minv[8];
  int mini[8];
#pragma unroll
  for (int v = 0; v < 8; ++v) { minv[v] = 3.4e38f; mini[v] = 0; }

  const int ncol = lane & 15;
  const int kbB = (lane & 16) ? 8 : 0;

  prefetch(0, 0);

  for (int ct = 0; ct < NITER; ++ct) {
    const int cur = ct & 1;
    if (ct + 1 < NITER) {
      prefetch(ct + 1, cur ^ 1);
      // Async loads complete in order: <=4 outstanding means buf[cur] is done.
      asm volatile("s_wait_asynccnt 0x4" ::: "memory");
    } else {
      asm volatile("s_wait_asynccnt 0x0" ::: "memory");
    }
    __syncthreads();

    const _Float16* cshi = (const _Float16*)(smem + csoff_hi[cur]);
    const _Float16* cslo = (const _Float16*)(smem + csoff_lo[cur]);

#pragma unroll
    for (int j = 0; j < CT / 16; ++j) {
      int lcol = j * 16 + ncol;
      // B 32x16 layout: col = lane&15; lanes 0-15 -> K {0..7,16..23}; lanes 16-31 -> +8.
      v16h bh[2], bl[2];
#pragma unroll
      for (int kkI = 0; kkI < 2; ++kkI) {
        int kb = kkI * 32 + kbB;
        v8h h0 = *(const v8h*)&cshi[lcol * RS + kb];
        v8h h1 = *(const v8h*)&cshi[lcol * RS + kb + 16];
        v8h l0 = *(const v8h*)&cslo[lcol * RS + kb];
        v8h l1 = *(const v8h*)&cslo[lcol * RS + kb + 16];
#pragma unroll
        for (int i = 0; i < 8; ++i) {
          bh[kkI][i] = h0[i]; bh[kkI][8 + i] = h1[i];
          bl[kkI][i] = l0[i]; bl[kkI][8 + i] = l1[i];
        }
      }
      // dot(x, c) ~= hi*hi + hi*lo + lo*hi  (f32 accumulate)
      v8f acc = {};
      acc = __builtin_amdgcn_wmma_f32_16x16x32_f16(false, axh[0], false, bh[0], (short)0, acc, false, false);
      acc = __builtin_amdgcn_wmma_f32_16x16x32_f16(false, axh[1], false, bh[1], (short)0, acc, false, false);
      acc = __builtin_amdgcn_wmma_f32_16x16x32_f16(false, axh[0], false, bl[0], (short)0, acc, false, false);
      acc = __builtin_amdgcn_wmma_f32_16x16x32_f16(false, axh[1], false, bl[1], (short)0, acc, false, false);
      acc = __builtin_amdgcn_wmma_f32_16x16x32_f16(false, axl[0], false, bh[0], (short)0, acc, false, false);
      acc = __builtin_amdgcn_wmma_f32_16x16x32_f16(false, axl[1], false, bh[1], (short)0, acc, false, false);

      int gcol = ct * CT + lcol;
      float cn = cns[cur][lcol];     // LDS broadcast, no global latency
      // C layout: VGPR v holds row (v + 8*(lane>=16)), col = lane&15.
#pragma unroll
      for (int v = 0; v < 8; ++v) {
        float dv = cn - 2.0f * acc[v];   // argmin over k of (||c||^2 - 2 x.c)
        if (dv < minv[v]) { minv[v] = dv; mini[v] = gcol; }
      }
    }
    __syncthreads();   // done reading buf[cur] before it is overwritten
  }

  // Reduce across the 16 column-lanes within each half-wave.
#pragma unroll
  for (int v = 0; v < 8; ++v) {
    float mv = minv[v];
    int mi = mini[v];
#pragma unroll
    for (int off = 8; off >= 1; off >>= 1) {
      float ov = __shfl_xor(mv, off, 32);
      int oi = __shfl_xor(mi, off, 32);
      if (ov < mv || (ov == mv && oi < mi)) { mv = ov; mi = oi; }
    }
    if ((lane & 15) == 0) {
      int grow = n0 + wv * 16 + ((lane & 16) ? 8 : 0) + v;
      idx[grow] = mi;
    }
  }
}

// ------------------------------------------------------------------
// Fused gather / straight-through output / segment sums / scalar sums.
__global__ void __launch_bounds__(256) vq_scatter(
    const float* __restrict__ x, const float* __restrict__ cb,
    const int* __restrict__ idx, float* __restrict__ out,
    float* __restrict__ cb_sum, float* __restrict__ counts,
    float* __restrict__ accum) {
  __shared__ float r0[256], r1[256], r2[256];
  size_t gi = ((size_t)blockIdx.x * 256 + threadIdx.x) * 4;
  int t = (int)(gi % TT);
  int e = (int)((gi / TT) % DIM);
  int b = (int)(gi / ((size_t)TT * DIM));
  const float4 xv = *(const float4*)&x[gi];
  int n = b * TT + t;
  const int4 iv = *(const int4*)&idx[n];
  float xs[4] = {xv.x, xv.y, xv.z, xv.w};
  int is[4] = {iv.x, iv.y, iv.z, iv.w};
  float sx = 0.f, sxx = 0.f, sd2 = 0.f;
  float os[4];
#pragma unroll
  for (int j = 0; j < 4; ++j) {
    float q = cb[(size_t)is[j] * DIM + e];
    float d = q - xs[j];
    os[j] = xs[j] + d;                 // out = x + sg(q - x), reference fp order
    sd2 += d * d;
    sx += xs[j];
    sxx += xs[j] * xs[j];
    atomicAdd(&cb_sum[(size_t)is[j] * DIM + e], xs[j]);
  }
  float4 ov; ov.x = os[0]; ov.y = os[1]; ov.z = os[2]; ov.w = os[3];
  *(float4*)&out[gi] = ov;
  if (e == 0) {
#pragma unroll
    for (int j = 0; j < 4; ++j) atomicAdd(&counts[is[j]], 1.0f);
  }
  r0[threadIdx.x] = sx; r1[threadIdx.x] = sxx; r2[threadIdx.x] = sd2;
  __syncthreads();
  for (int s = 128; s > 0; s >>= 1) {
    if ((int)threadIdx.x < s) {
      r0[threadIdx.x] += r0[threadIdx.x + s];
      r1[threadIdx.x] += r1[threadIdx.x + s];
      r2[threadIdx.x] += r2[threadIdx.x + s];
    }
    __syncthreads();
  }
  if (threadIdx.x == 0) {
    atomicAdd(&accum[0], r0[0]);
    atomicAdd(&accum[1], r1[0]);
    atomicAdd(&accum[2], r2[0]);
  }
}

// ------------------------------------------------------------------
// EMA codebook update, entropy, codebook diff.
__global__ void __launch_bounds__(256) vq_update(
    const float* __restrict__ x, const float* __restrict__ cb,
    const float* __restrict__ cb_avg, const float* __restrict__ vuse,
    const int* __restrict__ perm, const float* __restrict__ cb_sum,
    const float* __restrict__ counts, float* __restrict__ newcb,
    float* __restrict__ newavg, float* __restrict__ newuse,
    float* __restrict__ accum) {
  __shared__ float rr[256];
  int i = blockIdx.x * 256 + threadIdx.x;  // [0, K*D)
  int k = i / DIM, e = i % DIM;
  float avg = WDEC * cb_avg[i] + (1.0f - WDEC) * cb_sum[i];
  float usage = WDEC * vuse[k] + (1.0f - WDEC) * counts[k];
  int p = perm[k];
  int pb = p / TT, pt = p % TT;
  float rs = x[((size_t)pb * DIM + e) * TT + pt];   // xf[perm[k]][e]
  float ncb = (usage >= 1.0f) ? (avg / usage) : rs;
  newcb[i] = ncb;
  newavg[i] = avg;
  if (e == 0) newuse[k] = usage;
  float dd = ncb - cb[i];
  rr[threadIdx.x] = dd * dd;
  __syncthreads();
  for (int s = 128; s > 0; s >>= 1) {
    if ((int)threadIdx.x < s) rr[threadIdx.x] += rr[threadIdx.x + s];
    __syncthreads();
  }
  if (threadIdx.x == 0) atomicAdd(&accum[3], rr[0]);
  if (e == 0) {
    float pr = counts[k] * (1.0f / (float)NROW);  // sum(counts) == N exactly
    atomicAdd(&accum[4], pr * logf(pr + 1e-8f));
  }
}

// ------------------------------------------------------------------
__global__ void vq_finalize(const float* __restrict__ accum, float* __restrict__ out) {
  const float NE = (float)NROW * (float)DIM;
  float Sx = accum[0], Sxx = accum[1], Sd = accum[2], Scd = accum[3], Sent = accum[4];
  out[OUT_COMMIT] = Sd / NE;                      // commit_loss
  out[OUT_FIT] = Sd / (float)NROW;                // fit = mean min dist
  float var = Sxx - (Sx * Sx) / NE;
  if (var < 0.f) var = 0.f;
  out[OUT_XNORM] = sqrtf(var) / sqrtf(NE);        // ||x - mean|| / sqrt(NE)
  out[OUT_ENT] = -Sent;
  out[OUT_CBDIFF] = sqrtf(Scd) / sqrtf((float)(KCB * DIM));
}

// ------------------------------------------------------------------
extern "C" void kernel_launch(void* const* d_in, const int* in_sizes, int n_in,
                              void* d_out, int out_size, void* d_ws, size_t ws_size,
                              hipStream_t stream) {
  const float* x      = (const float*)d_in[0];
  const float* cb     = (const float*)d_in[1];
  const float* cb_avg = (const float*)d_in[2];
  const float* vuse   = (const float*)d_in[3];
  const int*   perm   = (const int*)d_in[4];
  float* out = (float*)d_out;

  char* ws = (char*)d_ws;
  _Float16* chi = (_Float16*)(ws + WS_CHI);
  _Float16* clo = (_Float16*)(ws + WS_CLO);
  float* cnorm  = (float*)(ws + WS_CNORM);
  int* idxp     = (int*)(ws + WS_IDX);
  float* cbsum  = (float*)(ws + WS_CBSUM);
  float* cnts   = (float*)(ws + WS_COUNTS);
  float* acc    = (float*)(ws + WS_ACCUM);

  vq_zero<<<(ZERO_FLOATS + 255) / 256, 256, 0, stream>>>((float*)(ws + WS_CBSUM),
                                                         ZERO_FLOATS);
  vq_prep_codebook<<<KCB / 8, 256, 0, stream>>>(cb, chi, clo, cnorm);
  vq_argmin<<<NROW / TM, 256, 0, stream>>>(x, chi, clo, cnorm, idxp);
  vq_scatter<<<(BB * DIM * TT) / (256 * 4), 256, 0, stream>>>(x, cb, idxp, out,
                                                              cbsum, cnts, acc);
  vq_update<<<(KCB * DIM) / 256, 256, 0, stream>>>(x, cb, cb_avg, vuse, perm,
                                                   cbsum, cnts, out + OUT_NEWCB,
                                                   out + OUT_NEWAVG,
                                                   out + OUT_NEWUSE, acc);
  vq_finalize<<<1, 1, 0, stream>>>(acc, out);
}
